// Graph_noConn_75196287418590
// MI455X (gfx1250) — matrix operand order repaired and checked by reading, not compiled, not run on hardware
//
#include <hip/hip_runtime.h>
#include <hip/hip_bf16.h>

typedef __attribute__((ext_vector_type(16))) _Float16 v16h;
typedef __attribute__((ext_vector_type(8)))  _Float16 v8h;
typedef __attribute__((ext_vector_type(8)))  float    v8f;

#define CDIV(a,b) (((a)+(b)-1)/(b))

// ---------------- float <-> order-preserving uint key (for atomic max) ------
__device__ __forceinline__ unsigned fenc(float f) {
    unsigned u = __float_as_uint(f);
    return (u & 0x80000000u) ? ~u : (u | 0x80000000u);
}
__device__ __forceinline__ float fdec(unsigned k) {
    unsigned u = (k & 0x80000000u) ? (k & 0x7FFFFFFFu) : ~k;
    return __uint_as_float(u);
}

// ---------------- fills ------------------------------------------------------
__global__ void fill_f32_k(float* p, float v, int n) {
    int i = blockIdx.x * blockDim.x + threadIdx.x;
    if (i < n) p[i] = v;
}
__global__ void fill_u32_k(unsigned* p, unsigned v, int n) {
    int i = blockIdx.x * blockDim.x + threadIdx.x;
    if (i < n) p[i] = v;
}

// ---------------- f32 -> f16 packing for WMMA --------------------------------
// A: [M,K] f32 row-major -> [Mpad,K32] f16 row-major, zero padded
__global__ void cvt_a_k(const float* __restrict__ X, _Float16* __restrict__ A,
                        int M, int K, int K32, int tot) {
    int i = blockIdx.x * blockDim.x + threadIdx.x;
    if (i >= tot) return;
    int r = i / K32, k = i - r * K32;
    A[i] = (r < M && k < K) ? (_Float16)X[(size_t)r * K + k] : (_Float16)0.0f;
}
// W: [K,N] f32 row-major -> [Npad64,K32] f16 (column-major by output col), zero padded
__global__ void cvt_w_k(const float* __restrict__ W, _Float16* __restrict__ Bt,
                        int K, int N, int K32, int tot) {
    int i = blockIdx.x * blockDim.x + threadIdx.x;
    if (i >= tot) return;
    int n = i / K32, k = i - n * K32;
    Bt[i] = (n < N && k < K) ? (_Float16)W[(size_t)k * N + n] : (_Float16)0.0f;
}

// ---------------- WMMA GEMM: C[M,N] = A*W (+bias)(+relu) --------------------
// one wave per 16x64 tile (A fragment reused across 4 WMMAs), 8 waves per block
__global__ void gemm_wmma_k(const _Float16* __restrict__ A, const _Float16* __restrict__ Bt,
                            const float* __restrict__ bias, float* __restrict__ C,
                            int M, int N, int K32, int tilesM, int tilesN64, int act) {
    int tile = blockIdx.x * blockDim.y + threadIdx.y;
    if (tile >= tilesM * tilesN64) return;
    int tm = tile / tilesN64, tn = tile - tm * tilesN64;
    int lane = threadIdx.x;
    int l15 = lane & 15, lhi = lane >> 4;
    // A frag lane layout: row = lane&15; lanes 0-15: K 0-15; lanes 16-31: K 16-31
    const _Float16* ap = A + (size_t)((tm << 4) + l15) * K32 + (lhi << 4);
    // B frag lane layout: col = lane&15; lanes 0-15: K 0-15; lanes 16-31: K 16-31
    const _Float16* bp0 = Bt + (size_t)((tn << 6) + l15) * K32 + (lhi << 4);
    size_t bstride = (size_t)16 * K32;     // 16 output columns apart
    v8f acc0 = {}, acc1 = {}, acc2 = {}, acc3 = {};
    for (int k = 0; k < K32; k += 32) {
        v8h alo = *(const v8h*)(ap + k);
        v8h ahi = *(const v8h*)(ap + k + 8);
        v16h af;
#pragma unroll
        for (int i = 0; i < 8; i++) { af[i] = alo[i]; af[i + 8] = ahi[i]; }
        v16h b0 = *(const v16h*)(bp0 + k);
        v16h b1 = *(const v16h*)(bp0 + bstride + k);
        v16h b2 = *(const v16h*)(bp0 + 2 * bstride + k);
        v16h b3 = *(const v16h*)(bp0 + 3 * bstride + k);
        acc0 = __builtin_amdgcn_wmma_f32_16x16x32_f16(false, af, false, b0, (short)0, acc0, false, false);
        acc1 = __builtin_amdgcn_wmma_f32_16x16x32_f16(false, af, false, b1, (short)0, acc1, false, false);
        acc2 = __builtin_amdgcn_wmma_f32_16x16x32_f16(false, af, false, b2, (short)0, acc2, false, false);
        acc3 = __builtin_amdgcn_wmma_f32_16x16x32_f16(false, af, false, b3, (short)0, acc3, false, false);
    }
    int rb = (tm << 4) + (lhi << 3);
    v8f accs[4] = {acc0, acc1, acc2, acc3};
#pragma unroll
    for (int t = 0; t < 4; t++) {
        int col = (tn << 6) + (t << 4) + l15;
        if (col < N) {
            float bv = bias ? bias[col] : 0.0f;
#pragma unroll
            for (int j = 0; j < 8; j++) {
                int row = rb + j;
                if (row < M) {
                    float v = accs[t][j] + bv;
                    if (act) v = fmaxf(v, 0.0f);
                    C[(size_t)row * N + col] = v;
                }
            }
        }
    }
}

// ---------------- GAT pieces -------------------------------------------------
__global__ void attn_coef_k(const float* __restrict__ H, const float* __restrict__ aS,
                            const float* __restrict__ aD, float* __restrict__ asrc,
                            float* __restrict__ adst, int Nn, int C) {
    int i = blockIdx.x * blockDim.x + threadIdx.x;
    if (i >= Nn * 2) return;
    int n = i >> 1, hd = i & 1;
    const float* hp = H + ((size_t)n * 2 + hd) * C;
    const float* s = aS + hd * C;
    const float* d = aD + hd * C;
    float s1 = 0.f, s2 = 0.f;
    for (int c = 0; c < C; c++) { float v = hp[c]; s1 += v * s[c]; s2 += v * d[c]; }
    asrc[i] = s1; adst[i] = s2;
}

__global__ void gat_edge_max_k(const int* __restrict__ src, const int* __restrict__ dst,
                               int E, int Nn, const float* __restrict__ asrc,
                               const float* __restrict__ adst, float* __restrict__ alpha,
                               unsigned* __restrict__ mk) {
    int i = blockIdx.x * blockDim.x + threadIdx.x;
    if (i >= (E + Nn) * 2) return;
    int e = i >> 1, hd = i & 1;
    int s = (e < E) ? src[e] : (e - E);
    int d = (e < E) ? dst[e] : (e - E);
    float v = asrc[s * 2 + hd] + adst[d * 2 + hd];
    v = (v > 0.f) ? v : 0.2f * v;           // leaky relu, slope 0.2
    alpha[i] = v;
    atomicMax(&mk[d * 2 + hd], fenc(v));
}

__global__ void gat_edge_exp_k(const int* __restrict__ dst, int E, int Nn,
                               float* __restrict__ alpha, const unsigned* __restrict__ mk,
                               float* __restrict__ ss) {
    int i = blockIdx.x * blockDim.x + threadIdx.x;
    if (i >= (E + Nn) * 2) return;
    int e = i >> 1, hd = i & 1;
    int d = (e < E) ? dst[e] : (e - E);
    float ex = expf(alpha[i] - fdec(mk[d * 2 + hd]));
    alpha[i] = ex;
    atomicAdd(&ss[d * 2 + hd], ex);
}

__global__ void gat_edge_norm_k(const int* __restrict__ dst, int E, int Nn,
                                float* __restrict__ alpha, const float* __restrict__ ss) {
    int i = blockIdx.x * blockDim.x + threadIdx.x;
    if (i >= (E + Nn) * 2) return;
    int e = i >> 1, hd = i & 1;
    int d = (e < E) ? dst[e] : (e - E);
    alpha[i] = alpha[i] / (ss[d * 2 + hd] + 1e-16f);
}

__global__ void gat_scatter_k(const int* __restrict__ src, const int* __restrict__ dst,
                              int E, int Nn, const float* __restrict__ alpha,
                              const float* __restrict__ H, float* __restrict__ Acc, int C) {
    int i = blockIdx.x * blockDim.x + threadIdx.x;
    if (i >= (E + Nn) * C) return;
    int e = i / C, c = i - e * C;
    int s = (e < E) ? src[e] : (e - E);
    int d = (e < E) ? dst[e] : (e - E);
    const float* hs = H + (size_t)s * 2 * C;
    float* ad = Acc + (size_t)d * 2 * C;
    atomicAdd(&ad[c],     alpha[e * 2 + 0] * hs[c]);
    atomicAdd(&ad[C + c], alpha[e * 2 + 1] * hs[C + c]);
}

__global__ void gat_final_k(const float* __restrict__ Acc, const float* __restrict__ b,
                            float* __restrict__ out, int Nn, int C, int relu) {
    int i = blockIdx.x * blockDim.x + threadIdx.x;
    if (i >= Nn * C) return;
    int n = i / C, c = i - n * C;
    const float* a = Acc + (size_t)n * 2 * C;
    float v = 0.5f * (a[c] + a[C + c]) + b[c];     // mean over 2 heads + bias
    if (relu) v = fmaxf(v, 0.f);
    out[i] = v;
}

// ---------------- GCN pieces -------------------------------------------------
__global__ void gcn_deg_k(const int* __restrict__ dst, const float* __restrict__ ew,
                          int E, int Nn, float* __restrict__ deg) {
    int i = blockIdx.x * blockDim.x + threadIdx.x;
    if (i >= E + Nn) return;
    int d = (i < E) ? dst[i] : (i - E);
    float w = (i < E) ? ew[i] : 1.0f;
    atomicAdd(&deg[d], w);
}
__global__ void gcn_dinv_k(const float* __restrict__ deg, float* __restrict__ dinv, int Nn) {
    int i = blockIdx.x * blockDim.x + threadIdx.x;
    if (i >= Nn) return;
    float d = deg[i];
    dinv[i] = (d > 0.f) ? rsqrtf(d) : 0.f;
}
__global__ void gcn_norm_k(const int* __restrict__ src, const int* __restrict__ dst,
                           const float* __restrict__ ew, int E, int Nn,
                           const float* __restrict__ dinv, float* __restrict__ norm) {
    int i = blockIdx.x * blockDim.x + threadIdx.x;
    if (i >= E + Nn) return;
    int s = (i < E) ? src[i] : (i - E);
    int d = (i < E) ? dst[i] : (i - E);
    float w = (i < E) ? ew[i] : 1.0f;
    norm[i] = dinv[s] * w * dinv[d];
}
__global__ void gcn_scatter_k(const int* __restrict__ src, const int* __restrict__ dst,
                              int E, int Nn, const float* __restrict__ norm,
                              const float* __restrict__ H, float* __restrict__ Acc, int C) {
    int i = blockIdx.x * blockDim.x + threadIdx.x;
    if (i >= (E + Nn) * C) return;
    int e = i / C, c = i - e * C;
    int s = (e < E) ? src[e] : (e - E);
    int d = (e < E) ? dst[e] : (e - E);
    atomicAdd(&Acc[(size_t)d * C + c], norm[e] * H[(size_t)s * C + c]);
}
__global__ void gcn_final_k(const float* __restrict__ Acc, const float* __restrict__ b,
                            float* __restrict__ out, int Nn, int C, int relu) {
    int i = blockIdx.x * blockDim.x + threadIdx.x;
    if (i >= Nn * C) return;
    int c = i % C;
    float v = Acc[i] + b[c];
    if (relu) v = fmaxf(v, 0.f);
    out[i] = v;
}

// ---------------- global max pool (segment max over batch) -------------------
__global__ void pool_max_k(const float* __restrict__ X, const int* __restrict__ batch,
                           int Nn, int C, unsigned* __restrict__ keys) {
    int i = blockIdx.x * blockDim.x + threadIdx.x;
    if (i >= Nn * C) return;
    int n = i / C, c = i - n * C;
    atomicMax(&keys[batch[n] * C + c], fenc(X[i]));
}
__global__ void keys_dec_k(const unsigned* __restrict__ keys, float* __restrict__ out, int n) {
    int i = blockIdx.x * blockDim.x + threadIdx.x;
    if (i < n) out[i] = fdec(keys[i]);
}

// ---------------- attention combiner -----------------------------------------
// score[b] = tanh(V[b]@W1 + b1) @ W2   (V:[256,128], W1:[128,64], W2:[64,1])
__global__ void proj_score_k(const float* __restrict__ V, const float* __restrict__ W1,
                             const float* __restrict__ b1, const float* __restrict__ W2,
                             float* __restrict__ score, int col) {
    int b = blockIdx.x, j = threadIdx.x;   // 64 threads
    float t = b1[j];
    for (int k = 0; k < 128; k++) t += V[b * 128 + k] * W1[k * 64 + j];
    t = tanhf(t) * W2[j];
    __shared__ float red[64];
    red[j] = t;
    __syncthreads();
    for (int off = 32; off > 0; off >>= 1) {
        if (j < off) red[j] += red[j + off];
        __syncthreads();
    }
    if (j == 0) score[b * 3 + col] = red[0];
}

__global__ void att_combine_k(const float* __restrict__ score, const float* __restrict__ xs,
                              float* __restrict__ emb) {
    int i = blockIdx.x * blockDim.x + threadIdx.x;
    if (i >= 256 * 128) return;
    int b = i >> 7, c = i & 127;
    float s0 = score[b * 3], s1 = score[b * 3 + 1], s2 = score[b * 3 + 2];
    float mx = fmaxf(s0, fmaxf(s1, s2));
    float e0 = expf(s0 - mx), e1 = expf(s1 - mx), e2 = expf(s2 - mx);
    float inv = 1.f / (e0 + e1 + e2);
    emb[b * 384 + c]       = e0 * inv * xs[i];
    emb[b * 384 + 128 + c] = e1 * inv * xs[256 * 128 + i];
    emb[b * 384 + 256 + c] = e2 * inv * xs[2 * 256 * 128 + i];
}

__global__ void final_out_k(const float* __restrict__ Z, const float* __restrict__ W,
                            const float* __restrict__ b, float* __restrict__ out) {
    int i = blockIdx.x * blockDim.x + threadIdx.x;
    if (i >= 256) return;
    float s = b[0];
    for (int k = 0; k < 512; k++) s += Z[i * 512 + k] * W[k];
    out[i] = s;
}

// =============================================================================
extern "C" void kernel_launch(void* const* d_in, const int* in_sizes, int n_in,
                              void* d_out, int out_size, void* d_ws, size_t ws_size,
                              hipStream_t stream) {
    (void)in_sizes; (void)n_in; (void)out_size; (void)ws_size;
    const int MOL_N = 50000, MOL_E = 200000;
    const int PRO_N = 80000, PRO_E = 400000;
    const int CLQ_N = 40000, CLQ_E = 160000;

    auto f  = [&](int i) { return (const float*)d_in[i]; };
    auto ip = [&](int i) { return (const int*)d_in[i]; };

    // ---- workspace bump allocator ----
    char* wsBase = (char*)d_ws;
    size_t wsOff = 0;
    auto walloc = [&](size_t bytes) -> void* {
        wsOff = (wsOff + 255) & ~(size_t)255;
        void* p = wsBase + wsOff;
        wsOff += bytes;
        return p;
    };
    float*    F_H    = (float*)walloc(31200000ull * 4);  // max N x 2C (mol: 50000x624)
    float*    F_ACC  = (float*)walloc(31200000ull * 4);
    float*    P0     = (float*)walloc(15600000ull * 4);  // max N x C
    float*    P1     = (float*)walloc(15600000ull * 4);
    _Float16* A16    = (_Float16*)walloc(16000000ull * 2);
    _Float16* W16    = (_Float16*)walloc(600000ull * 2);
    float*    F_ASRC = (float*)walloc(160000 * 4);
    float*    F_ADST = (float*)walloc(160000 * 4);
    unsigned* U_MK   = (unsigned*)walloc(160000 * 4);
    float*    F_SS   = (float*)walloc(160000 * 4);
    float*    F_ALPHA= (float*)walloc(960000 * 4);
    float*    F_DEG  = (float*)walloc(80000 * 4);
    float*    F_DINV = (float*)walloc(80000 * 4);
    float*    F_NORM = (float*)walloc(480000 * 4);
    unsigned* U_PK   = (unsigned*)walloc(94208 * 4);     // 256 x 368 max
    float*    F_POOL = (float*)walloc(94208 * 4);
    float*    F_G1   = (float*)walloc(262144 * 4);       // 256 x 1024
    float*    F_X    = (float*)walloc(98304 * 4);        // [3][256][128]
    float*    F_SC   = (float*)walloc(768 * 4);
    float*    F_EMB  = (float*)walloc(98304 * 4);        // 256 x 384
    float*    F_Z1   = (float*)walloc(262144 * 4);
    float*    F_Z2   = (float*)walloc(131072 * 4);

    // ---- WMMA GEMM wrapper: Cout[M,N] = X[M,K]@W[K,N] (+bias)(+relu) ----
    auto gemm = [&](const float* X, int M, int K, const float* W, int N,
                    const float* bias, float* Cout, int act) {
        int K32 = CDIV(K, 32) * 32;
        int Mpad = CDIV(M, 16) * 16;
        int Npad64 = CDIV(N, 64) * 64;
        int totA = Mpad * K32;
        cvt_a_k<<<CDIV(totA, 256), 256, 0, stream>>>(X, A16, M, K, K32, totA);
        int totB = Npad64 * K32;
        cvt_w_k<<<CDIV(totB, 256), 256, 0, stream>>>(W, W16, K, N, K32, totB);
        int tiles = (Mpad / 16) * (Npad64 / 64);
        gemm_wmma_k<<<CDIV(tiles, 8), dim3(32, 8), 0, stream>>>(
            A16, W16, bias, Cout, M, N, K32, Mpad / 16, Npad64 / 64, act);
    };

    // ---- one GAT layer ----
    auto gat_layer = [&](const float* Xin, int Nn, int Fin, int E,
                         const int* src, const int* dst,
                         const float* W, const float* aS, const float* aD,
                         const float* bias, int C, float* out, int relu) {
        gemm(Xin, Nn, Fin, W, 2 * C, nullptr, F_H, 0);
        attn_coef_k<<<CDIV(Nn * 2, 256), 256, 0, stream>>>(F_H, aS, aD, F_ASRC, F_ADST, Nn, C);
        fill_u32_k<<<CDIV(Nn * 2, 256), 256, 0, stream>>>(U_MK, 0u, Nn * 2);
        fill_f32_k<<<CDIV(Nn * 2, 256), 256, 0, stream>>>(F_SS, 0.f, Nn * 2);
        int Et2 = (E + Nn) * 2;
        gat_edge_max_k<<<CDIV(Et2, 256), 256, 0, stream>>>(src, dst, E, Nn, F_ASRC, F_ADST, F_ALPHA, U_MK);
        gat_edge_exp_k<<<CDIV(Et2, 256), 256, 0, stream>>>(dst, E, Nn, F_ALPHA, U_MK, F_SS);
        gat_edge_norm_k<<<CDIV(Et2, 256), 256, 0, stream>>>(dst, E, Nn, F_ALPHA, F_SS);
        fill_f32_k<<<CDIV(Nn * 2 * C, 256), 256, 0, stream>>>(F_ACC, 0.f, Nn * 2 * C);
        int totS = (E + Nn) * C;
        gat_scatter_k<<<CDIV(totS, 256), 256, 0, stream>>>(src, dst, E, Nn, F_ALPHA, F_H, F_ACC, C);
        gat_final_k<<<CDIV(Nn * C, 256), 256, 0, stream>>>(F_ACC, bias, out, Nn, C, relu);
    };

    // ---- pooled FC head: gmp -> fc1(relu,1024) -> fc2(128) ----
    auto head = [&](const float* Hf, int Nn, int C, const int* batch,
                    const float* W1, const float* b1, const float* W2, const float* b2,
                    float* xout) {
        fill_u32_k<<<CDIV(256 * C, 256), 256, 0, stream>>>(U_PK, 0u, 256 * C);
        pool_max_k<<<CDIV(Nn * C, 256), 256, 0, stream>>>(Hf, batch, Nn, C, U_PK);
        keys_dec_k<<<CDIV(256 * C, 256), 256, 0, stream>>>(U_PK, F_POOL, 256 * C);
        gemm(F_POOL, 256, C, W1, 1024, b1, F_G1, 1);
        gemm(F_G1, 256, 1024, W2, 128, b2, xout, 0);
    };

    // ================= molecule branch =================
    const int* msrc = ip(62); const int* mdst = msrc + MOL_E;
    gat_layer(f(61), MOL_N, 78,  MOL_E, msrc, mdst, f(0), f(1), f(2),  f(3),  312, P0, 1);
    gat_layer(P0,    MOL_N, 312, MOL_E, msrc, mdst, f(4), f(5), f(6),  f(7),  312, P1, 1);
    gat_layer(P1,    MOL_N, 312, MOL_E, msrc, mdst, f(8), f(9), f(10), f(11), 312, P0, 0);
    head(P0, MOL_N, 312, ip(63), f(34), f(35), f(36), f(37), F_X);

    // ================= protein branch =================
    const int* psrc = ip(65); const int* pdst = psrc + PRO_E;
    const float* pew = f(66);
    gemm(f(64), PRO_N, 33, f(12), 132, nullptr, F_H, 0);
    fill_f32_k<<<CDIV(PRO_N, 256), 256, 0, stream>>>(F_DEG, 0.f, PRO_N);
    gcn_deg_k<<<CDIV(PRO_E + PRO_N, 256), 256, 0, stream>>>(pdst, pew, PRO_E, PRO_N, F_DEG);
    gcn_dinv_k<<<CDIV(PRO_N, 256), 256, 0, stream>>>(F_DEG, F_DINV, PRO_N);
    gcn_norm_k<<<CDIV(PRO_E + PRO_N, 256), 256, 0, stream>>>(psrc, pdst, pew, PRO_E, PRO_N, F_DINV, F_NORM);
    fill_f32_k<<<CDIV(PRO_N * 132, 256), 256, 0, stream>>>(F_ACC, 0.f, PRO_N * 132);
    gcn_scatter_k<<<CDIV((PRO_E + PRO_N) * 132, 256), 256, 0, stream>>>(
        psrc, pdst, PRO_E, PRO_N, F_NORM, F_H, F_ACC, 132);
    gcn_final_k<<<CDIV(PRO_N * 132, 256), 256, 0, stream>>>(F_ACC, f(13), P0, PRO_N, 132, 1);
    gat_layer(P0, PRO_N, 132, PRO_E, psrc, pdst, f(14), f(15), f(16), f(17), 132, P1, 1);
    gat_layer(P1, PRO_N, 132, PRO_E, psrc, pdst, f(18), f(19), f(20), f(21), 132, P0, 0);
    head(P0, PRO_N, 132, ip(67), f(38), f(39), f(40), f(41), F_X + 256 * 128);

    // ================= clique branch =================
    const int* csrc = ip(69); const int* cdst = csrc + CLQ_E;
    gat_layer(f(68), CLQ_N, 92,  CLQ_E, csrc, cdst, f(22), f(23), f(24), f(25), 368, P0, 1);
    gat_layer(P0,    CLQ_N, 368, CLQ_E, csrc, cdst, f(26), f(27), f(28), f(29), 368, P1, 1);
    gat_layer(P1,    CLQ_N, 368, CLQ_E, csrc, cdst, f(30), f(31), f(32), f(33), 368, P0, 0);
    head(P0, CLQ_N, 368, ip(70), f(42), f(43), f(44), f(45), F_X + 2 * 256 * 128);

    // ================= attention combiner + final MLP =================
    proj_score_k<<<256, 64, 0, stream>>>(F_X,                 f(46), f(47), f(48), F_SC, 0);
    proj_score_k<<<256, 64, 0, stream>>>(F_X + 256 * 128,     f(49), f(50), f(51), F_SC, 1);
    proj_score_k<<<256, 64, 0, stream>>>(F_X + 2 * 256 * 128, f(52), f(53), f(54), F_SC, 2);
    att_combine_k<<<CDIV(256 * 128, 256), 256, 0, stream>>>(F_SC, F_X, F_EMB);
    gemm(F_EMB, 256, 384,  f(55), 1024, f(56), F_Z1, 1);
    gemm(F_Z1,  256, 1024, f(57), 512,  f(58), F_Z2, 1);
    final_out_k<<<1, 256, 0, stream>>>(F_Z2, f(59), f(60), (float*)d_out);
}
